// GraphAttentionLayer_77919296684444
// MI455X (gfx1250) — matrix operand order repaired and checked by reading, not compiled, not run on hardware
//
#include <hip/hip_runtime.h>
#include <hip/hip_bf16.h>

typedef __attribute__((ext_vector_type(2)))  float    v2f;
typedef __attribute__((ext_vector_type(8)))  float    v8f;
typedef __attribute__((ext_vector_type(16))) _Float16 v16h;

#define NB   8
#define NN   2048
#define NF   64
#define ALPHA 0.2f

// ---------------- workspace layout (bytes) ----------------
#define WS_H     ((size_t)0)              // 8*2048*64 f32   = 4 MiB
#define WS_S1    ((size_t)4194304)        // 16384 f32
#define WS_S2    ((size_t)4259840)
#define WS_M     ((size_t)4325376)
#define WS_BITS  ((size_t)4456448)        // 2048*64 u32 = 512 KiB
#define WS_HB    ((size_t)4980736)        // 8*64*4*32*8 dwords = 2 MiB (f16 swizzled h)

// =====================================================================
// K1: h = input @ W  (f32 WMMA 16x16x4), fused s1 = h.a1, s2 = h.a2
// One wave per 16-row tile of the flattened (B*N, 64) h matrix.
// A-frag f32 16x4:  lane L: M=L%16; VGPR0 K=(L<16?0:2), VGPR1 K=+1
// B-frag f32 4x16:  lane L: N=L%16; VGPR0 K=(L<16?0:2), VGPR1 K=+1
// C 16x16:          VGPR r: lanes0-15 M=r, lanes16-31 M=8+r, N=L%16
// =====================================================================
__global__ __launch_bounds__(256) void k_gemm_h(
    const float* __restrict__ inp, const float* __restrict__ W,
    const float* __restrict__ a,
    float* __restrict__ h, float* __restrict__ s1, float* __restrict__ s2) {
  const int lane = threadIdx.x & 31;
  const int wv   = threadIdx.x >> 5;
  const int tile = blockIdx.x * 8 + wv;          // 1024 tiles
  const int r0   = tile * 16;
  const int m    = lane & 15;
  const int hi   = lane >> 4;

  v8f acc[4] = {v8f{0}, v8f{0}, v8f{0}, v8f{0}};
  const float* arow = inp + (size_t)(r0 + m) * NF;

  for (int k0 = 0; k0 < NF; k0 += 4) {
    v2f A = *(const v2f*)(arow + k0 + 2 * hi);
#pragma unroll
    for (int g = 0; g < 4; ++g) {
      v2f Bf;
      Bf.x = W[(k0 + 2 * hi)     * NF + g * 16 + m];
      Bf.y = W[(k0 + 2 * hi + 1) * NF + g * 16 + m];
      acc[g] = __builtin_amdgcn_wmma_f32_16x16x4_f32(
          false, A, false, Bf, (short)0, acc[g], false, false);
    }
  }

  float a1v[4], a2v[4];
#pragma unroll
  for (int g = 0; g < 4; ++g) { a1v[g] = a[g * 16 + m]; a2v[g] = a[NF + g * 16 + m]; }

#pragma unroll
  for (int r = 0; r < 8; ++r) {
    const int row = r0 + r + 8 * hi;
    float t1 = 0.f, t2 = 0.f;
#pragma unroll
    for (int g = 0; g < 4; ++g) {
      float v = acc[g][r];
      h[(size_t)row * NF + g * 16 + m] = v;
      t1 += v * a1v[g];
      t2 += v * a2v[g];
    }
    // reduce over the 16 lanes of this half (N dimension)
#pragma unroll
    for (int off = 1; off < 16; off <<= 1) {
      t1 += __shfl_xor(t1, off, 32);
      t2 += __shfl_xor(t2, off, 32);
    }
    if (m == 0) { s1[row] = t1; s2[row] = t2; }
  }
}

// =====================================================================
// K2: adj (N x N int32) -> per-row 32-bit masks. One wave per row.
// =====================================================================
__global__ __launch_bounds__(256) void k_bits(
    const int* __restrict__ adj, unsigned* __restrict__ bits) {
  const int lane = threadIdx.x & 31;
  const int wv   = threadIdx.x >> 5;
  const int row  = blockIdx.x * 8 + wv;          // 2048 rows
  const int* arow = adj + (size_t)row * NN;
  for (int w = 0; w < NN / 32; ++w) {
    int v = arow[w * 32 + lane];
    unsigned msk = __builtin_amdgcn_ballot_w32(v > 0);
    if (lane == 0) bits[row * (NN / 32) + w] = msk;
  }
}

// =====================================================================
// K3: per (b,i): M = lrelu(s1 + max_{valid j} s2[j])
// (leaky-ReLU is monotonic, so masked row-max = lrelu(s1 + masked max s2))
// No exp pass here: the softmax denominator is accumulated for free inside
// the main WMMA kernel from the P fragment values. One wave per row.
// =====================================================================
__global__ __launch_bounds__(256) void k_rowmax(
    const unsigned* __restrict__ bits, const float* __restrict__ s1,
    const float* __restrict__ s2, float* __restrict__ Mrow) {
  const int lane = threadIdx.x & 31;
  const int wv   = threadIdx.x >> 5;
  const int row  = blockIdx.x * 8 + wv;          // 16384 rows
  const int b    = row >> 11;
  const int i    = row & (NN - 1);
  const unsigned* br  = bits + i * (NN / 32);
  const float*    s2b = s2 + b * NN;

  float mx = -3.0e38f;
  for (int w = 0; w < NN / 32; ++w) {
    unsigned msk = br[w];
    float v = s2b[w * 32 + lane];
    if ((msk >> lane) & 1u) mx = fmaxf(mx, v);
  }
#pragma unroll
  for (int off = 1; off < 32; off <<= 1) mx = fmaxf(mx, __shfl_xor(mx, off, 32));

  float e = s1[row] + mx;
  float M = e > 0.f ? e : ALPHA * e;
  if (mx < -2.9e38f) M = 0.f;                    // fully-masked row fallback

  if (lane == 0) Mrow[row] = M;
}

// =====================================================================
// K4: repack h (f32 row-major) -> HB: f16 pre-swizzled WMMA-B fragments.
// B-frag 16-bit 32x16: VGPR v, N=L%16: K=2*(v%4)+16*(v>=4)+8*(L>=16) +{0,1}
// Flat dword index: b*65536 + c*1024 + g*256 + L*8 + v
// so the main kernel reads each fragment as 2 coalesced b128 loads per lane.
// =====================================================================
__global__ __launch_bounds__(256) void k_pack(
    const float* __restrict__ h, unsigned* __restrict__ HB) {
  const unsigned t = blockIdx.x * 256 + threadIdx.x;   // 524288 dwords
  const int v  = t & 7;
  const int L  = (t >> 3) & 31;
  const int g  = (t >> 8) & 3;
  const int c  = (t >> 10) & 63;
  const int bb = t >> 16;
  const int k0  = 2 * (v & 3) + ((v >> 2) << 4) + ((L >> 4) << 3);
  const int col = g * 16 + (L & 15);
  const int j   = c * 32 + k0;
  const float* base = h + ((size_t)bb * NN + j) * NF + col;
  union { _Float16 hh[2]; unsigned u; } pk;
  pk.hh[0] = (_Float16)base[0];
  pk.hh[1] = (_Float16)base[NF];
  HB[t] = pk.u;
}

// =====================================================================
// K5: fused masked-softmax * V matmul + ELU.
// One wave per (b, 16-row tile). Per 32-col chunk c:
//   - synthesize A (P weights, f16) directly in the WMMA A-layout:
//     lane L: M=L%16; VGPR p holds K = 2*(p%4) + 16*(p>=4) + 8*(L>=16) +{0,1}
//   - accumulate the softmax denominator from the same f32 P values
//     (row m lives only in lanes m and m+16 -> one shfl_xor(16) merge)
//   - 4x v_wmma_f32_16x16x32_f16 into 16x64 f32 accumulators
// =====================================================================
__global__ __launch_bounds__(256) void k_attn(
    const unsigned* __restrict__ bits, const float* __restrict__ s1,
    const float* __restrict__ s2, const float* __restrict__ Mrow,
    const uint4* __restrict__ HB, float* __restrict__ out) {
  const int lane = threadIdx.x & 31;
  const int wv   = threadIdx.x >> 5;
  const int b    = blockIdx.y;
  const int tile = blockIdx.x * 8 + wv;          // 128 tiles
  const int i0   = tile * 16;
  const int m    = lane & 15;
  const int hi   = lane >> 4;

  const float s1m = s1[b * NN + i0 + m];
  const float Mm  = Mrow[b * NN + i0 + m];
  const unsigned* br  = bits + (i0 + m) * (NN / 32);
  const float*    s2b = s2 + b * NN;
  const uint4*    hbB = HB + (size_t)b * 16384;  // 65536 dwords / 4

  v8f acc[4] = {v8f{0}, v8f{0}, v8f{0}, v8f{0}};
  float psum = 0.f;                              // lane-local sum of P values

  for (int c = 0; c < NN / 32; ++c) {
    const unsigned msk = br[c];
    union { v16h v; unsigned w[8]; } A;
#pragma unroll
    for (int p = 0; p < 8; ++p) {
      const int base = 2 * (p & 3) + ((p >> 2) << 4) + (hi << 3);
      v2f sv = *(const v2f*)(s2b + c * 32 + base);
      float e0 = s1m + sv.x; e0 = e0 > 0.f ? e0 : ALPHA * e0;
      float e1 = s1m + sv.y; e1 = e1 > 0.f ? e1 : ALPHA * e1;
      float p0 = ((msk >> base)       & 1u) ? __expf(e0 - Mm) : 0.f;
      float p1 = ((msk >> (base + 1)) & 1u) ? __expf(e1 - Mm) : 0.f;
      psum += p0 + p1;
      union { _Float16 hh[2]; unsigned u; } pk;
      pk.hh[0] = (_Float16)p0;
      pk.hh[1] = (_Float16)p1;
      A.w[p] = pk.u;
    }
#pragma unroll
    for (int g = 0; g < 4; ++g) {
      union { v16h v; uint4 q[2]; } B;
      const uint4* pB = hbB + c * 256 + g * 64 + lane * 2;
      B.q[0] = pB[0];
      B.q[1] = pB[1];
      acc[g] = __builtin_amdgcn_wmma_f32_16x16x32_f16(
          false, A.v, false, B.v, (short)0, acc[g], false, false);
    }
  }

  // merge the two K-halves of each row: lanes m and m+16 both hold row m
  psum += __shfl_xor(psum, 16, 32);

  // epilogue: scale by 1/rowsum, ELU, store.
  // C layout row = r + 8*hi; its sum lives in lane (r + 8*hi).
#pragma unroll
  for (int r = 0; r < 8; ++r) {
    const int row = i0 + r + 8 * hi;
    const float s  = __shfl(psum, r + 8 * hi, 32);
    const float si = (s > 0.f) ? (1.f / s) : 0.f;
#pragma unroll
    for (int g = 0; g < 4; ++g) {
      float x = acc[g][r] * si;
      x = x > 0.f ? x : (__expf(x) - 1.f);
      out[((size_t)(b * NN + row)) * NF + g * 16 + m] = x;
    }
  }
}

extern "C" void kernel_launch(void* const* d_in, const int* in_sizes, int n_in,
                              void* d_out, int out_size, void* d_ws, size_t ws_size,
                              hipStream_t stream) {
  const float* inp = (const float*)d_in[0];   // (8, 2048, 64) f32
  const int*   adj = (const int*)d_in[1];     // (2048, 2048) i32
  const float* W   = (const float*)d_in[2];   // (64, 64) f32
  const float* a   = (const float*)d_in[3];   // (128, 1) f32
  float* out = (float*)d_out;                 // (8, 2048, 64) f32

  char* ws = (char*)d_ws;
  float*    h    = (float*)(ws + WS_H);
  float*    s1   = (float*)(ws + WS_S1);
  float*    s2   = (float*)(ws + WS_S2);
  float*    Mrow = (float*)(ws + WS_M);
  unsigned* bits = (unsigned*)(ws + WS_BITS);
  unsigned* HB   = (unsigned*)(ws + WS_HB);

  k_gemm_h<<<dim3(128), dim3(256), 0, stream>>>(inp, W, a, h, s1, s2);
  k_bits<<<dim3(256), dim3(256), 0, stream>>>(adj, bits);
  k_rowmax<<<dim3(2048), dim3(256), 0, stream>>>(bits, s1, s2, Mrow);
  k_pack<<<dim3(2048), dim3(256), 0, stream>>>(h, HB);
  k_attn<<<dim3(16, 8), dim3(256), 0, stream>>>(bits, s1, s2, Mrow,
                                                (const uint4*)HB, out);
}
